// MultiHeadAttention_25013889532363
// MI455X (gfx1250) — matrix operand order repaired and checked by reading, not compiled, not run on hardware
//
#include <hip/hip_runtime.h>
#include <hip/hip_bf16.h>

// MI455X / gfx1250, wave32. Heavy math via v_wmma_f32_16x16x32_bf16,
// K-tile staging via TDM tensor_load_to_lds, double-buffered so the DMA of
// tile i+1 overlaps the WMMA+softmax of tile i (s_wait_tensorcnt fences).

typedef __bf16 bf16;
typedef __attribute__((ext_vector_type(16))) __bf16 v16bf;
typedef __attribute__((ext_vector_type(8)))  __bf16 v8bf;
typedef __attribute__((ext_vector_type(8)))  float  v8f;
typedef __attribute__((ext_vector_type(4)))  unsigned int u32x4;
typedef __attribute__((ext_vector_type(8)))  int i32x8;
typedef __attribute__((ext_vector_type(4)))  int i32x4;

#define D_MODEL 1024
#define SEQ     2048
#define NHEAD   16
#define DK      64
#define NKT     (SEQ / 32)

#if defined(__has_builtin)
#if __has_builtin(__builtin_amdgcn_tensor_load_to_lds) && \
    __has_builtin(__builtin_amdgcn_s_wait_tensorcnt)
#define HAVE_TDM 1
#endif
#endif
#ifndef HAVE_TDM
#define HAVE_TDM 0
#endif

// ---------------------------------------------------------------- convert
__global__ __launch_bounds__(256) void f32_to_bf16_kernel(
    const float* __restrict__ src, bf16* __restrict__ dst, int n8) {
  int i = blockIdx.x * 256 + threadIdx.x;
  if (i < n8) {
    const float4* s = (const float4*)src + (size_t)i * 2;
    float4 f0 = s[0], f1 = s[1];
    v8bf o;
    o[0] = (bf16)f0.x; o[1] = (bf16)f0.y; o[2] = (bf16)f0.z; o[3] = (bf16)f0.w;
    o[4] = (bf16)f1.x; o[5] = (bf16)f1.y; o[6] = (bf16)f1.z; o[7] = (bf16)f1.w;
    *((v8bf*)dst + i) = o;
  }
}

// A-fragment loader: 16x32 bf16, row-major source, contiguous k order (k = 16*hi + s)
__device__ __forceinline__ v16bf load_a_frag(const bf16* __restrict__ row,
                                             int k, int hi) {
  v8bf lo = *(const v8bf*)(row + k + 16 * hi);
  v8bf hv = *(const v8bf*)(row + k + 16 * hi + 8);
  v16bf a;
#pragma unroll
  for (int i = 0; i < 8; ++i) { a[i] = lo[i]; a[i + 8] = hv[i]; }
  return a;
}

// ------------------------------------------------------------ QKV GEMM
// 32x64 tile per wave: 2 A-frags share every B-frag (2x reuse).
__global__ __launch_bounds__(256) void gemm_qkv_kernel(
    const bf16* __restrict__ xb, const bf16* __restrict__ wb,
    const float* __restrict__ bias,
    bf16* __restrict__ Qg, bf16* __restrict__ Kg, bf16* __restrict__ Vg) {
  const int tid = threadIdx.x;
  const int wv = tid >> 5, lane = tid & 31, l16 = lane & 15, hi = lane >> 4;
  const int wid = blockIdx.x * 8 + wv;        // 6144 wave tiles
  const int mTile = wid / 48;                 // 0..127 (32 rows each)
  const int nCh   = wid % 48;                 // 0..47  (64 cols each)

  const bf16* xrow0 = xb + (size_t)(mTile * 32 + l16) * D_MODEL;
  const bf16* xrow1 = xrow0 + (size_t)16 * D_MODEL;
  v8f acc[2][4] = {};

  for (int k = 0; k < D_MODEL; k += 32) {
    __builtin_prefetch(xrow0 + k + 256, 0, 1);   // global_prefetch_b8
    __builtin_prefetch(xrow1 + k + 256, 0, 1);
    v16bf a0 = load_a_frag(xrow0, k, hi);
    v16bf a1 = load_a_frag(xrow1, k, hi);
#pragma unroll
    for (int nt = 0; nt < 4; ++nt) {
      int n = nCh * 64 + nt * 16 + l16;
      v16bf b = *(const v16bf*)(wb + (size_t)n * D_MODEL + k + 16 * hi);
      acc[0][nt] = __builtin_amdgcn_wmma_f32_16x16x32_bf16(
          false, a0, false, b, (short)0, acc[0][nt], false, false);
      acc[1][nt] = __builtin_amdgcn_wmma_f32_16x16x32_bf16(
          false, a1, false, b, (short)0, acc[1][nt], false, false);
    }
  }

  const int sel = (nCh * 64) >> 10;  // 0=Q 1=K 2=V
  bf16* outT = sel == 0 ? Qg : (sel == 1 ? Kg : Vg);
#pragma unroll
  for (int half = 0; half < 2; ++half)
#pragma unroll
    for (int nt = 0; nt < 4; ++nt) {
      int n = nCh * 64 + nt * 16 + l16;
      int d = n & (D_MODEL - 1);
      int h = d >> 6, dkk = d & 63;
      float bv = bias[n];
#pragma unroll
      for (int r = 0; r < 8; ++r) {
        int m = mTile * 32 + half * 16 + r + 8 * hi;
        int b = m >> 11, s = m & (SEQ - 1);
        outT[((size_t)(b * NHEAD + h) * SEQ + s) * DK + dkk] =
            (bf16)(acc[half][nt][r] + bv);
      }
    }
}

// ------------------------------------------------------- flash attention
// grid (S/64, B*H), 128 threads (4 waves); wave handles 16 query rows.
// K tiles via TDM, double-buffered; V tiles transposed into LDS by the block.
__global__ __launch_bounds__(128) void mha_flash_kernel(
    const bf16* __restrict__ Qg, const bf16* __restrict__ Kg,
    const bf16* __restrict__ Vg, bf16* __restrict__ ctx) {
  __shared__ bf16 Klds[2][32][80];    // K chunk: 128B data + 32B pad per row
  __shared__ bf16 Vt[2][64][48];      // V chunk transposed [dk][key]
  __shared__ bf16 Plds[4][16][48];    // per-wave P tile [qrow][key]

  const int tid = threadIdx.x;
  const int wv = tid >> 5, lane = tid & 31, l16 = lane & 15, hi = lane >> 4;
  const int bh = blockIdx.y;
  const int qblk = blockIdx.x;

  const bf16* Qbh = Qg + (size_t)bh * SEQ * DK;
  const bf16* Kbh = Kg + (size_t)bh * SEQ * DK;
  const bf16* Vbh = Vg + (size_t)bh * SEQ * DK;

  const int qrow = qblk * 64 + wv * 16 + l16;

#if HAVE_TDM
  // wave-uniform TDM DMA of one K tile (32 rows x 64 bf16) into Klds[buf]
  auto issue_k_tdm = [&](int kt, int buf) {
    unsigned lds_off = (unsigned)(unsigned long long)(const void*)&Klds[buf][0][0];
    unsigned long long ga =
        (unsigned long long)(const void*)(Kbh + (size_t)kt * DK);
    u32x4 g0;
    g0[0] = 1u;                                   // count=1, user mode
    g0[1] = lds_off;                              // lds_addr
    g0[2] = (unsigned)(ga & 0xFFFFFFFFu);         // global_addr lo
    g0[3] = (unsigned)((ga >> 32) & 0x01FFFFFFu) | (2u << 30);  // addr hi | type=2
    i32x8 g1;
    g1[0] = (1 << 16)   // data_size = 2 bytes
          | (1 << 20)   // pad_enable
          | (4 << 22)   // pad_interval: 2^(4+1)=32 DWORDs (=128B row)
          | (7 << 25);  // pad_amount: 7+1=8 DWORDs (=32B pad -> 160B stride)
    g1[1] = (DK & 0xFFFF) << 16;                  // tensor_dim0 = 64 (lo)
    g1[2] = ((SEQ & 0xFFFF) << 16);               // dim0 hi=0 | tensor_dim1 lo
    g1[3] = (DK << 16);                           // dim1 hi=0 | tile_dim0 = 64
    g1[4] = 32;                                   // tile_dim1 = 32, tile_dim2 = 0
    g1[5] = DK;                                   // tensor_dim0_stride = 64
    g1[6] = 0;
    g1[7] = 0;
    i32x4 gz = {0, 0, 0, 0};
#if __clang_major__ >= 23
    i32x8 gz8 = {0, 0, 0, 0, 0, 0, 0, 0};
    __builtin_amdgcn_tensor_load_to_lds(g0, g1, gz, gz, gz8, 0);
#else
    __builtin_amdgcn_tensor_load_to_lds(g0, g1, gz, gz, 0);
#endif
  };
#else
  auto load_k_fallback = [&](int kt, int buf) {
    int row = tid >> 2, seg = tid & 3;
    v16bf kv = *(const v16bf*)(Kbh + (size_t)(kt + row) * DK + seg * 16);
    *(v16bf*)&Klds[buf][row][seg * 16] = kv;
  };
#endif
  auto load_v = [&](int kt, int buf) {   // all 128 threads: load + transpose
    int row = tid >> 2, seg = tid & 3;
    v16bf vv = *(const v16bf*)(Vbh + (size_t)(kt + row) * DK + seg * 16);
#pragma unroll
    for (int i = 0; i < 16; ++i) Vt[buf][seg * 16 + i][row] = vv[i];
  };

  v16bf aq[2];
#pragma unroll
  for (int c = 0; c < 2; ++c)
    aq[c] = load_a_frag(Qbh + (size_t)qrow * DK, c * 32, hi);

  float mrow[8], lrow[8];
#pragma unroll
  for (int r = 0; r < 8; ++r) { mrow[r] = -3.0e38f; lrow[r] = 0.f; }
  v8f oacc[4] = {};

  const float SCL = 0.125f * 1.44269504f;  // 1/sqrt(64) * log2(e)

  // ---- prologue: start tile 0 into buffer 0
#if HAVE_TDM
  if (wv == 0) issue_k_tdm(0, 0);
#else
  load_k_fallback(0, 0);
#endif
  load_v(0, 0);

  for (int it = 0; it < NKT; ++it) {
    const int cur = it & 1;
#if HAVE_TDM
    if (wv == 0) __builtin_amdgcn_s_wait_tensorcnt(0);  // tile 'it' K in LDS
#endif
    __syncthreads();  // publish K/V of tile 'it'; all reads of buf 1-cur (it-1) done

    if (it + 1 < NKT) {   // start tile it+1 into the other buffer (overlaps compute)
#if HAVE_TDM
      if (wv == 0) issue_k_tdm((it + 1) * 32, 1 - cur);
#else
      load_k_fallback((it + 1) * 32, 1 - cur);
#endif
      load_v((it + 1) * 32, 1 - cur);
    }

    // scores: two 16-key tiles, dk reduced in 2 WMMA steps each
    v8f sc[2] = {};
#pragma unroll
    for (int t = 0; t < 2; ++t)
#pragma unroll
      for (int c = 0; c < 2; ++c) {
        v16bf bk = *(const v16bf*)&Klds[cur][t * 16 + l16][c * 32 + 16 * hi];
        sc[t] = __builtin_amdgcn_wmma_f32_16x16x32_bf16(
            false, aq[c], false, bk, (short)0, sc[t], false, false);
      }

    // online softmax (rows r+8*hi live in the 16-lane half-groups)
    float corr[8];
#pragma unroll
    for (int r = 0; r < 8; ++r) {
      float s0 = sc[0][r] * SCL, s1 = sc[1][r] * SCL;
      float v = fmaxf(s0, s1);
#pragma unroll
      for (int msk = 8; msk >= 1; msk >>= 1) v = fmaxf(v, __shfl_xor(v, msk, 32));
      float nm = fmaxf(mrow[r], v);
      corr[r] = exp2f(mrow[r] - nm);
      mrow[r] = nm;
      float p0 = exp2f(s0 - nm), p1 = exp2f(s1 - nm);
      float ps = p0 + p1;
#pragma unroll
      for (int msk = 8; msk >= 1; msk >>= 1) ps += __shfl_xor(ps, msk, 32);
      lrow[r] = lrow[r] * corr[r] + ps;
      Plds[wv][r + 8 * hi][l16]      = (bf16)p0;
      Plds[wv][r + 8 * hi][16 + l16] = (bf16)p1;
    }
#pragma unroll
    for (int td = 0; td < 4; ++td)
#pragma unroll
      for (int r = 0; r < 8; ++r) oacc[td][r] *= corr[r];

    asm volatile("s_wait_dscnt 0" ::: "memory");  // per-wave LDS RAW on Plds

    v16bf ap;   // A fragment of P (16 q-rows x 32 keys)
    {
      const bf16* pr = &Plds[wv][l16][0];
      v8bf lo = *(const v8bf*)(pr + 16 * hi);
      v8bf hv = *(const v8bf*)(pr + 16 * hi + 8);
#pragma unroll
      for (int i = 0; i < 8; ++i) { ap[i] = lo[i]; ap[i + 8] = hv[i]; }
    }
#pragma unroll
    for (int td = 0; td < 4; ++td) {
      v16bf bv = *(const v16bf*)&Vt[cur][td * 16 + l16][16 * hi];
      oacc[td] = __builtin_amdgcn_wmma_f32_16x16x32_bf16(
          false, ap, false, bv, (short)0, oacc[td], false, false);
    }
  }

  // normalize & write context [B,S,D] bf16 (head-major columns)
  const int b = bh >> 4, h = bh & 15;
#pragma unroll
  for (int td = 0; td < 4; ++td)
#pragma unroll
    for (int r = 0; r < 8; ++r) {
      int row = qblk * 64 + wv * 16 + r + 8 * hi;
      float o = oacc[td][r] / lrow[r];
      ctx[(size_t)(b * SEQ + row) * D_MODEL + h * 64 + td * 16 + l16] = (bf16)o;
    }
}

// --------------------------------------------------------- output GEMM
__global__ __launch_bounds__(256) void gemm_out_kernel(
    const bf16* __restrict__ cb, const bf16* __restrict__ wb,
    const float* __restrict__ bias, float* __restrict__ out) {
  const int tid = threadIdx.x;
  const int wv = tid >> 5, lane = tid & 31, l16 = lane & 15, hi = lane >> 4;
  const int wid = blockIdx.x * 8 + wv;        // 2048 wave tiles
  const int mTile = wid / 16;                 // 0..127 (32 rows each)
  const int nCh   = wid % 16;

  const bf16* crow0 = cb + (size_t)(mTile * 32 + l16) * D_MODEL;
  const bf16* crow1 = crow0 + (size_t)16 * D_MODEL;
  v8f acc[2][4] = {};

  for (int k = 0; k < D_MODEL; k += 32) {
    __builtin_prefetch(crow0 + k + 256, 0, 1);
    __builtin_prefetch(crow1 + k + 256, 0, 1);
    v16bf a0 = load_a_frag(crow0, k, hi);
    v16bf a1 = load_a_frag(crow1, k, hi);
#pragma unroll
    for (int nt = 0; nt < 4; ++nt) {
      int n = nCh * 64 + nt * 16 + l16;
      v16bf b = *(const v16bf*)(wb + (size_t)n * D_MODEL + k + 16 * hi);
      acc[0][nt] = __builtin_amdgcn_wmma_f32_16x16x32_bf16(
          false, a0, false, b, (short)0, acc[0][nt], false, false);
      acc[1][nt] = __builtin_amdgcn_wmma_f32_16x16x32_bf16(
          false, a1, false, b, (short)0, acc[1][nt], false, false);
    }
  }
#pragma unroll
  for (int half = 0; half < 2; ++half)
#pragma unroll
    for (int nt = 0; nt < 4; ++nt) {
      int n = nCh * 64 + nt * 16 + l16;
      float bv = bias[n];
#pragma unroll
      for (int r = 0; r < 8; ++r) {
        int m = mTile * 32 + half * 16 + r + 8 * hi;
        out[(size_t)m * D_MODEL + n] = acc[half][nt][r] + bv;
      }
    }
}

// ---------------------------------------------------------------- launch
extern "C" void kernel_launch(void* const* d_in, const int* in_sizes, int n_in,
                              void* d_out, int out_size, void* d_ws, size_t ws_size,
                              hipStream_t stream) {
  const float* x     = (const float*)d_in[0];   // [2,2048,1024]
  const float* w_qkv = (const float*)d_in[1];   // [3072,1024]
  const float* b_qkv = (const float*)d_in[2];   // [3072]
  const float* w_out = (const float*)d_in[3];   // [1024,1024]
  const float* b_out = (const float*)d_in[4];   // [1024]
  float* out = (float*)d_out;                   // [2,2048,1024]

  const size_t N_X   = (size_t)2 * SEQ * D_MODEL;        // 4194304
  const size_t N_WQ  = (size_t)3 * D_MODEL * D_MODEL;    // 3145728
  const size_t N_WO  = (size_t)D_MODEL * D_MODEL;        // 1048576
  const size_t N_HSD = (size_t)2 * NHEAD * SEQ * DK;     // 4194304

  bf16* xb    = (bf16*)d_ws;
  bf16* wqkvb = xb + N_X;
  bf16* woutb = wqkvb + N_WQ;
  bf16* Qg    = woutb + N_WO;
  bf16* Kg    = Qg + N_HSD;
  bf16* Vg    = Kg + N_HSD;
  bf16* ctx   = Vg + N_HSD;                     // total ~50 MB

  f32_to_bf16_kernel<<<(int)(N_X  / 8 / 256), 256, 0, stream>>>(x, xb, (int)(N_X / 8));
  f32_to_bf16_kernel<<<(int)(N_WQ / 8 / 256), 256, 0, stream>>>(w_qkv, wqkvb, (int)(N_WQ / 8));
  f32_to_bf16_kernel<<<(int)(N_WO / 8 / 256), 256, 0, stream>>>(w_out, woutb, (int)(N_WO / 8));

  gemm_qkv_kernel<<<768, 256, 0, stream>>>(xb, wqkvb, b_qkv, Qg, Kg, Vg);

  dim3 g3(SEQ / 64, 2 * NHEAD);
  mha_flash_kernel<<<g3, 128, 0, stream>>>(Qg, Kg, Vg, ctx);

  gemm_out_kernel<<<256, 256, 0, stream>>>(ctx, woutb, b_out, out);
}
